// NodeModel_80032420593874
// MI455X (gfx1250) — matrix operand order repaired and checked by reading, not compiled, not run on hardware
//
#include <hip/hip_runtime.h>

// ---------------- problem constants (match reference) ----------------
#define N_NODES  100000   // divisible by 32 -> no tail handling in fused kernel
#define N_EDGES  1600000
#define D_X      128
#define D_E      64
#define D_U      64
#define D_IN     256   // D_X + D_E + D_U
#define D_HID    256
#define D_OUT    128

// ---------------- WMMA types (CDNA5 gfx1250, wave32) ----------------
typedef __bf16        v16bf __attribute__((ext_vector_type(16)));
typedef float         v8f   __attribute__((ext_vector_type(8)));
typedef unsigned int  v4u   __attribute__((ext_vector_type(4)));

union Frag {            // one 16x32 (A) or 32x16 (B) bf16 fragment per lane:
    v16bf bf;           // 16 bf16 = 8 VGPRs
    v4u   u2[2];        // loaded as two 16-byte vectors
};

__device__ __forceinline__ unsigned short f2bf(float f) {
    unsigned int u = __float_as_uint(f);
    u += 0x7FFFu + ((u >> 16) & 1u);      // round-to-nearest-even
    return (unsigned short)(u >> 16);
}

// ---------------- kernel 1: zero the agg scratch ----------------
__global__ void zero_agg_kernel(float4* __restrict__ p, int n4) {
    int i = blockIdx.x * blockDim.x + threadIdx.x;
    if (i < n4) p[i] = make_float4(0.f, 0.f, 0.f, 0.f);
}

// ---------------- kernel 2: scatter-add edge features ----------------
// one thread per (edge, 4-dim chunk): float4 load + 4 fp32 global atomics
__global__ void scatter_edges_kernel(const float* __restrict__ edge_attr,
                                     const int*   __restrict__ edge_row, // edge_index[0][:]
                                     float*       __restrict__ agg) {
    int t = blockIdx.x * blockDim.x + threadIdx.x;     // [0, N_EDGES*16)
    int e = t >> 4;
    int c = (t & 15) * 4;
    if (e >= N_EDGES) return;
    int row = edge_row[e];
    float4 v = *(const float4*)(edge_attr + (size_t)e * D_E + c);
    float* dst = agg + (size_t)row * D_E + c;
    unsafeAtomicAdd(dst + 0, v.x);                     // global_atomic_add_f32
    unsafeAtomicAdd(dst + 1, v.y);
    unsafeAtomicAdd(dst + 2, v.z);
    unsafeAtomicAdd(dst + 3, v.w);
}

// ---------------- kernel 3: pack W1/W2 -> bf16 B-fragments ----------------
// B fragment (32x16 bf16): lane L holds column n = L%16; half indices h=0..15
// map to K = 8*(L>=16) + h for h<8, 16 + 8*(L>=16) + (h-8) for h>=8,
// stored contiguously: frag*512 + lane*16 + h  halfs.
__global__ void pack_weights_kernel(const float* __restrict__ W1,
                                    const float* __restrict__ W2,
                                    unsigned short* __restrict__ W1f,
                                    unsigned short* __restrict__ W2f) {
    int t = blockIdx.x * blockDim.x + threadIdx.x;     // 192 frags * 32 lanes
    int f = t >> 5, lane = t & 31;
    if (f >= 192) return;
    int n  = lane & 15;
    int hi = (lane >> 4) & 1;
    if (f < 128) {                                     // W1: 8 k-tiles x 16 n-tiles
        int kt = f >> 4, nt = f & 15;
        unsigned short* dst = W1f + (size_t)f * 512 + lane * 16;
        #pragma unroll
        for (int h = 0; h < 16; ++h) {
            int K = kt * 32 + (h & 7) + hi * 8 + (h >> 3) * 16;
            dst[h] = f2bf(W1[(size_t)K * D_HID + nt * 16 + n]);
        }
    } else {                                           // W2: 8 k-tiles x 8 n-tiles
        int g = f - 128;
        int kt = g >> 3, nt = g & 7;
        unsigned short* dst = W2f + (size_t)g * 512 + lane * 16;
        #pragma unroll
        for (int h = 0; h < 16; ++h) {
            int K = kt * 32 + (h & 7) + hi * 8 + (h >> 3) * 16;
            dst[h] = f2bf(W2[(size_t)K * D_OUT + nt * 16 + n]);
        }
    }
}

// ---------------- kernel 4: fused gather + 2-layer MLP via WMMA ----------------
// 32 nodes per block (two 16-row M-tiles); 256 threads = 8 wave32.
// Each B fragment fetched once per block now feeds TWO WMMAs (both M-tiles),
// halving L2 weight traffic per FLOP vs a 16-row tile.
// LDS rows padded to 272 halfs (544 B, 16B multiple) for b128 alignment and
// bank-conflict avoidance.  N_NODES % 32 == 0 -> no bounds checks, EXEC
// trivially all-ones around every WMMA.
__global__ __launch_bounds__(256)
void fused_mlp_kernel(const float* __restrict__ x,
                      const float* __restrict__ agg,
                      const float* __restrict__ u,
                      const int*   __restrict__ v_indices,
                      const unsigned short* __restrict__ W1f,
                      const unsigned short* __restrict__ W2f,
                      const float* __restrict__ b1,
                      const float* __restrict__ b2,
                      float* __restrict__ out) {
    __shared__ __align__(16) unsigned short hA[32][272];  // input tile  [32 x 256] bf16
    __shared__ __align__(16) unsigned short hH[32][272];  // hidden tile [32 x 256] bf16

    const int tid   = threadIdx.x;
    const int node0 = blockIdx.x * 32;

    // ---- stage h = [x | agg | u[v_idx]] tile into LDS as bf16 ----
    // 32 rows x 256 cols; thread t -> row t/8, 32-col group t%8.
    // Region boundaries (128, 192) are multiples of 32, so each thread's
    // group lies entirely in one source tensor.
    {
        int r = tid >> 3;             // row 0..31
        int g = tid & 7;              // 32-col group
        int node = node0 + r;
        int vi = v_indices[node];
        #pragma unroll
        for (int q = 0; q < 8; ++q) {
            int c = g * 32 + q * 4;
            float4 v;
            if (c < D_X)            v = *(const float4*)(x   + (size_t)node * D_X + c);
            else if (c < D_X + D_E) v = *(const float4*)(agg + (size_t)node * D_E + (c - D_X));
            else                    v = *(const float4*)(u   + (size_t)vi   * D_U + (c - D_X - D_E));
            hA[r][c + 0] = f2bf(v.x);
            hA[r][c + 1] = f2bf(v.y);
            hA[r][c + 2] = f2bf(v.z);
            hA[r][c + 3] = f2bf(v.w);
        }
    }
    __syncthreads();

    const int lane = tid & 31;
    const int wave = tid >> 5;
    const int hi   = (lane >> 4) & 1;   // half-wave select
    const int n    = lane & 15;         // column within 16-wide tile / row M for A

    // ---- GEMM1: h[32,256] @ W1[256,256] + b1, ReLU
    // wave owns 2 hidden col-tiles x 2 M-tiles = 4 accumulators
    {
        const int nt0 = wave * 2, nt1 = wave * 2 + 1;
        v8f acc00, acc01, acc10, acc11;        // [mtile][ntile]
        float bv0 = b1[nt0 * 16 + n];
        float bv1 = b1[nt1 * 16 + n];
        #pragma unroll
        for (int r = 0; r < 8; ++r) {
            acc00[r] = bv0; acc01[r] = bv1;
            acc10[r] = bv0; acc11[r] = bv1;
        }

        #pragma unroll
        for (int kt = 0; kt < 8; ++kt) {
            // A fragments: rows n (M-tile 0) and n+16 (M-tile 1);
            // halfs h<8 -> K = kt*32 + hi*8 + h, h>=8 -> +16
            Frag a0, a1;
            const unsigned short* ap0 = &hA[n     ][kt * 32 + hi * 8];
            const unsigned short* ap1 = &hA[n + 16][kt * 32 + hi * 8];
            a0.u2[0] = *(const v4u*)(ap0);          // ds_load_b128
            a0.u2[1] = *(const v4u*)(ap0 + 16);
            a1.u2[0] = *(const v4u*)(ap1);
            a1.u2[1] = *(const v4u*)(ap1 + 16);

            Frag bA, bB;
            const unsigned short* bp0 = W1f + ((size_t)(kt * 16 + nt0)) * 512 + lane * 16;
            const unsigned short* bp1 = W1f + ((size_t)(kt * 16 + nt1)) * 512 + lane * 16;
            bA.u2[0] = *(const v4u*)(bp0);          // global_load_b128 (L2-resident)
            bA.u2[1] = *(const v4u*)(bp0 + 8);
            bB.u2[0] = *(const v4u*)(bp1);
            bB.u2[1] = *(const v4u*)(bp1 + 8);

            acc00 = __builtin_amdgcn_wmma_f32_16x16x32_bf16(
                        false, a0.bf, false, bA.bf, (short)0, acc00, false, false);
            acc01 = __builtin_amdgcn_wmma_f32_16x16x32_bf16(
                        false, a0.bf, false, bB.bf, (short)0, acc01, false, false);
            acc10 = __builtin_amdgcn_wmma_f32_16x16x32_bf16(
                        false, a1.bf, false, bA.bf, (short)0, acc10, false, false);
            acc11 = __builtin_amdgcn_wmma_f32_16x16x32_bf16(
                        false, a1.bf, false, bB.bf, (short)0, acc11, false, false);
        }

        // ReLU + write hidden tiles (C/D layout: m = r + 8*hi, col = nt*16 + n)
        #pragma unroll
        for (int r = 0; r < 8; ++r) {
            int m = r + hi * 8;
            hH[m     ][nt0 * 16 + n] = f2bf(fmaxf(acc00[r], 0.f));
            hH[m     ][nt1 * 16 + n] = f2bf(fmaxf(acc01[r], 0.f));
            hH[m + 16][nt0 * 16 + n] = f2bf(fmaxf(acc10[r], 0.f));
            hH[m + 16][nt1 * 16 + n] = f2bf(fmaxf(acc11[r], 0.f));
        }
    }
    __syncthreads();

    // ---- GEMM2: hid[32,256] @ W2[256,128] + b2; wave owns 1 col-tile x 2 M-tiles
    {
        const int nt = wave;            // 8 waves x 16 cols = 128
        v8f acc0, acc1;
        float bv = b2[nt * 16 + n];
        #pragma unroll
        for (int r = 0; r < 8; ++r) { acc0[r] = bv; acc1[r] = bv; }

        #pragma unroll
        for (int kt = 0; kt < 8; ++kt) {
            Frag a0, a1;
            const unsigned short* ap0 = &hH[n     ][kt * 32 + hi * 8];
            const unsigned short* ap1 = &hH[n + 16][kt * 32 + hi * 8];
            a0.u2[0] = *(const v4u*)(ap0);
            a0.u2[1] = *(const v4u*)(ap0 + 16);
            a1.u2[0] = *(const v4u*)(ap1);
            a1.u2[1] = *(const v4u*)(ap1 + 16);

            Frag b;
            const unsigned short* bp = W2f + ((size_t)(kt * 8 + nt)) * 512 + lane * 16;
            b.u2[0] = *(const v4u*)(bp);
            b.u2[1] = *(const v4u*)(bp + 8);

            acc0 = __builtin_amdgcn_wmma_f32_16x16x32_bf16(
                       false, a0.bf, false, b.bf, (short)0, acc0, false, false);
            acc1 = __builtin_amdgcn_wmma_f32_16x16x32_bf16(
                       false, a1.bf, false, b.bf, (short)0, acc1, false, false);
        }

        #pragma unroll
        for (int r = 0; r < 8; ++r) {
            int m = r + hi * 8;
            out[(size_t)(node0 + m     ) * D_OUT + nt * 16 + n] = acc0[r];
            out[(size_t)(node0 + m + 16) * D_OUT + nt * 16 + n] = acc1[r];
        }
    }
}

// ---------------- launcher ----------------
extern "C" void kernel_launch(void* const* d_in, const int* in_sizes, int n_in,
                              void* d_out, int out_size, void* d_ws, size_t ws_size,
                              hipStream_t stream) {
    const float* x    = (const float*)d_in[0];
    const int*   ei   = (const int*)  d_in[1];   // (2, E); row 0 first
    const float* ea   = (const float*)d_in[2];
    const float* u    = (const float*)d_in[3];
    const int*   vidx = (const int*)  d_in[4];
    const float* W1   = (const float*)d_in[5];
    const float* b1   = (const float*)d_in[6];
    const float* W2   = (const float*)d_in[7];
    const float* b2   = (const float*)d_in[8];
    float* out = (float*)d_out;

    // workspace layout
    char* ws = (char*)d_ws;
    const size_t AGG_BYTES = (size_t)N_NODES * D_E * sizeof(float);      // 25.6 MB
    float*          agg = (float*)ws;
    unsigned short* W1f = (unsigned short*)(ws + AGG_BYTES);             // 128 frags * 1KB
    unsigned short* W2f = W1f + 128 * 512;                               //  64 frags * 1KB

    // 1) zero agg
    {
        int n4 = (N_NODES * D_E) / 4;                    // 1.6M float4
        int blocks = (n4 + 255) / 256;
        zero_agg_kernel<<<blocks, 256, 0, stream>>>((float4*)agg, n4);
    }
    // 2) pack weights to bf16 fragments
    {
        int threads = 192 * 32;
        pack_weights_kernel<<<(threads + 255) / 256, 256, 0, stream>>>(W1, W2, W1f, W2f);
    }
    // 3) scatter-add edge features
    {
        long long t = (long long)N_EDGES * 16;
        int blocks = (int)((t + 255) / 256);             // 100,000
        scatter_edges_kernel<<<blocks, 256, 0, stream>>>(ea, ei, agg);
    }
    // 4) fused gather + MLP (WMMA), 32 nodes/block
    {
        int blocks = N_NODES / 32;                       // 3125
        fused_mlp_kernel<<<blocks, 256, 0, stream>>>(x, agg, u, vidx, W1f, W2f, b1, b2, out);
    }
}
